// KSparseAutoencoder_90958817395430
// MI455X (gfx1250) — matrix operand order repaired and checked by reading, not compiled, not run on hardware
//
#include <hip/hip_runtime.h>

typedef __attribute__((ext_vector_type(16))) __bf16 v16bf;
typedef __attribute__((ext_vector_type(8)))  __bf16 v8bf;
typedef __attribute__((ext_vector_type(8)))  float  v8f;
typedef __attribute__((ext_vector_type(4)))  unsigned int v4u;
typedef __attribute__((ext_vector_type(8)))  int    v8i;
typedef __attribute__((ext_vector_type(4)))  int    v4i;

constexpr int N_ROWS  = 8192;
constexpr int D_MODEL = 1024;
constexpr int NDIRS   = 16384;
constexpr int TOPK    = 32;
constexpr int MULTIK  = 128;

// -------------------- precision conversion (fold pre_bias into x) --------------------
__global__ void convert_x_kernel(const float* __restrict__ x,
                                 const float* __restrict__ pre_bias,
                                 __bf16* __restrict__ xbf) {
  int i = blockIdx.x * blockDim.x + threadIdx.x;
  int d = i & (D_MODEL - 1);
  xbf[i] = (__bf16)(x[i] - pre_bias[d]);
}

__global__ void convert_w_kernel(const float* __restrict__ w, __bf16* __restrict__ wbf) {
  int i = blockIdx.x * blockDim.x + threadIdx.x;
  wbf[i] = (__bf16)w[i];
}

// -------------------- W_dec [D, NDIRS] -> W_decT [NDIRS, D] (fp32) --------------------
__global__ void transpose_wdec_kernel(const float* __restrict__ Wd, float* __restrict__ WdT) {
  __shared__ float tile[32][33];
  int c  = blockIdx.x * 32 + threadIdx.x;  // NDIRS dim
  int r0 = blockIdx.y * 32;                // D dim
#pragma unroll
  for (int j = 0; j < 4; ++j) {
    int r = r0 + threadIdx.y + j * 8;
    tile[threadIdx.y + j * 8][threadIdx.x] = Wd[(size_t)r * NDIRS + c];
  }
  __syncthreads();
  int co  = r0 + threadIdx.x;              // D dim (out col)
  int ro0 = blockIdx.x * 32;               // NDIRS dim (out row)
#pragma unroll
  for (int j = 0; j < 4; ++j) {
    int ro = ro0 + threadIdx.y + j * 8;
    WdT[(size_t)ro * D_MODEL + co] = tile[threadIdx.x][threadIdx.y + j * 8];
  }
}

// -------------------- coalesced zero of the latents output --------------------
__global__ void zero_latents_kernel(float4* __restrict__ p) {
  size_t i = (size_t)blockIdx.x * blockDim.x + threadIdx.x;
  p[i] = make_float4(0.f, 0.f, 0.f, 0.f);
}

// -------------------- encoder GEMM: TDM tile DMA + bf16 WMMA, fp32 accumulate --------
constexpr int LDS_STRIDE = 40;  // halfwords per 32-elem row (64B data + 16B pad)
constexpr int KT = D_MODEL / 32;

#define USE_TDM (__has_builtin(__builtin_amdgcn_tensor_load_to_lds) && \
                 __has_builtin(__builtin_amdgcn_s_wait_tensorcnt))

#if USE_TDM
// Issue one TDM descriptor load: 128x32 bf16 tile (row-major, row stride 1024 elems)
// -> LDS with 16B padding appended after every 64B row (LDS row stride 80B).
__device__ __forceinline__ void tdm_load_tile(const __bf16* gsrc, unsigned lds_addr,
                                              unsigned tensor_dim1) {
  unsigned long long ga = (unsigned long long)(uintptr_t)gsrc;
  // D# group0: count=1 | lds_addr | global_addr[56:0] | type=2
  v4u g0 = { 1u,
             lds_addr,
             (unsigned)(ga & 0xFFFFFFFFu),
             (unsigned)((ga >> 32) & 0x01FFFFFFu) | 0x80000000u };
  // D# group1: data_size=1(2B) | pad_enable | pad_interval=3(16 DW) | pad_amount=3(4 DW)
  //            tensor_dim0=1024, tensor_dim1, tile_dim0=32, tile_dim1=128,
  //            tensor_dim0_stride=1024
  v8i g1 = { (int)0x06D10000u,
             (int)(1024u << 16),
             (int)(tensor_dim1 << 16),
             (int)(32u << 16),
             128,
             1024,
             0, 0 };
  v4i z4 = { 0, 0, 0, 0 };
#if __clang_major__ >= 23
  v8i z8 = { 0, 0, 0, 0, 0, 0, 0, 0 };
  __builtin_amdgcn_tensor_load_to_lds(g0, g1, z4, z4, z8, 0);
#else
  __builtin_amdgcn_tensor_load_to_lds(g0, g1, z4, z4, 0);
#endif
}
#endif

__device__ __forceinline__ v16bf load_frag(const __bf16* base, int o0, int o1) {
  v8bf lo = *(const v8bf*)(base + o0);
  v8bf hi = *(const v8bf*)(base + o1);
  v16bf f;
#pragma unroll
  for (int i = 0; i < 8; ++i) { f[i] = lo[i]; f[i + 8] = hi[i]; }
  return f;
}

__global__ void __launch_bounds__(256) encoder_gemm_kernel(
    const __bf16* __restrict__ xbf, const __bf16* __restrict__ wbf,
    const float* __restrict__ latent_bias, float* __restrict__ lat_pre) {
  __shared__ __align__(16) __bf16 As[2][128 * LDS_STRIDE];
  __shared__ __align__(16) __bf16 Bs[2][128 * LDS_STRIDE];

  const int t    = threadIdx.x;
  const int m0   = blockIdx.y * 128;
  const int n0   = blockIdx.x * 128;
  const int wave = t >> 5;
  const int lane = t & 31;
  const int hs   = (lane >> 4) & 1;  // lane half selector
  const int r16  = lane & 15;
  const int wm   = wave & 3;   // M quarter (32 rows)
  const int wn   = wave >> 2;  // N half   (64 cols)

  v8f acc[2][4] = {};

#if USE_TDM
  // Prologue: wave 0 DMAs tile 0 into buffer 0, everyone waits at the barrier.
  if (t < 32) {
    tdm_load_tile(&xbf[(size_t)m0 * D_MODEL], (unsigned)(uintptr_t)&As[0][0], N_ROWS);
    tdm_load_tile(&wbf[(size_t)n0 * D_MODEL], (unsigned)(uintptr_t)&Bs[0][0], NDIRS);
    __builtin_amdgcn_s_wait_tensorcnt(0);
  }
  __syncthreads();

  for (int kt = 0; kt < KT; ++kt) {
    const int cur = kt & 1, nxt = cur ^ 1;
    // Overlap: DMA next K-slab into the other buffer while we do WMMAs on this one.
    if (t < 32 && kt + 1 < KT) {
      tdm_load_tile(&xbf[(size_t)m0 * D_MODEL + (kt + 1) * 32],
                    (unsigned)(uintptr_t)&As[nxt][0], N_ROWS);
      tdm_load_tile(&wbf[(size_t)n0 * D_MODEL + (kt + 1) * 32],
                    (unsigned)(uintptr_t)&Bs[nxt][0], NDIRS);
    }

    // A 16x32 bf16 layout: lanes0-15 hold K{0-7,16-23}, lanes16-31 K{8-15,24-31}
    v16bf afrag[2];
#pragma unroll
    for (int am = 0; am < 2; ++am) {
      const int row = wm * 32 + am * 16 + r16;
      afrag[am] = load_frag(&As[cur][row * LDS_STRIDE], hs * 8, hs * 8 + 16);
    }
    // B 32x16 bf16 layout: lanes0-15 hold K{0-7,16-23}, lanes16-31 K{8-15,24-31}
    v16bf bfrag[4];
#pragma unroll
    for (int bn = 0; bn < 4; ++bn) {
      const int row = wn * 64 + bn * 16 + r16;
      bfrag[bn] = load_frag(&Bs[cur][row * LDS_STRIDE], hs * 8, hs * 8 + 16);
    }
#pragma unroll
    for (int am = 0; am < 2; ++am)
#pragma unroll
      for (int bn = 0; bn < 4; ++bn)
        acc[am][bn] = __builtin_amdgcn_wmma_f32_16x16x32_bf16(
            false, afrag[am], false, bfrag[bn], (short)0, acc[am][bn], false, false);

    if (t < 32 && kt + 1 < KT) __builtin_amdgcn_s_wait_tensorcnt(0);
    __syncthreads();
  }
#else
  // Fallback: per-thread VGPR staging (proven path).
  const int c0 = t, c1 = t + 256;
  const int r_a0 = c0 >> 2, q_a0 = c0 & 3;
  const int r_a1 = c1 >> 2, q_a1 = c1 & 3;
  for (int kt = 0; kt < KT; ++kt) {
    const int k0 = kt * 32;
    *(v8bf*)&As[0][r_a0 * LDS_STRIDE + q_a0 * 8] =
        *(const v8bf*)&xbf[(size_t)(m0 + r_a0) * D_MODEL + k0 + q_a0 * 8];
    *(v8bf*)&As[0][r_a1 * LDS_STRIDE + q_a1 * 8] =
        *(const v8bf*)&xbf[(size_t)(m0 + r_a1) * D_MODEL + k0 + q_a1 * 8];
    *(v8bf*)&Bs[0][r_a0 * LDS_STRIDE + q_a0 * 8] =
        *(const v8bf*)&wbf[(size_t)(n0 + r_a0) * D_MODEL + k0 + q_a0 * 8];
    *(v8bf*)&Bs[0][r_a1 * LDS_STRIDE + q_a1 * 8] =
        *(const v8bf*)&wbf[(size_t)(n0 + r_a1) * D_MODEL + k0 + q_a1 * 8];
    __syncthreads();
    v16bf afrag[2];
#pragma unroll
    for (int am = 0; am < 2; ++am) {
      const int row = wm * 32 + am * 16 + r16;
      afrag[am] = load_frag(&As[0][row * LDS_STRIDE], hs * 8, hs * 8 + 16);
    }
    v16bf bfrag[4];
#pragma unroll
    for (int bn = 0; bn < 4; ++bn) {
      const int row = wn * 64 + bn * 16 + r16;
      bfrag[bn] = load_frag(&Bs[0][row * LDS_STRIDE], hs * 8, hs * 8 + 16);
    }
#pragma unroll
    for (int am = 0; am < 2; ++am)
#pragma unroll
      for (int bn = 0; bn < 4; ++bn)
        acc[am][bn] = __builtin_amdgcn_wmma_f32_16x16x32_bf16(
            false, afrag[am], false, bfrag[bn], (short)0, acc[am][bn], false, false);
    __syncthreads();
  }
#endif

  float bias[4];
#pragma unroll
  for (int bn = 0; bn < 4; ++bn) bias[bn] = latent_bias[n0 + wn * 64 + bn * 16 + r16];

  // C/D layout: VGPR r -> M = r + hs*8, lane -> N = r16
#pragma unroll
  for (int am = 0; am < 2; ++am) {
    const int row_base = m0 + wm * 32 + am * 16 + hs * 8;
#pragma unroll
    for (int bn = 0; bn < 4; ++bn) {
      const int col = n0 + wn * 64 + bn * 16 + r16;
#pragma unroll
      for (int r = 0; r < 8; ++r)
        lat_pre[(size_t)(row_base + r) * NDIRS + col] = acc[am][bn][r] + bias[bn];
    }
  }
}

// -------------------- per-row top-128 (and top-32 prefix) via LDS argmax-extract ------
__global__ void __launch_bounds__(256) topk_kernel(
    const float* __restrict__ lat_pre, float* __restrict__ latents,
    float* __restrict__ topv, int* __restrict__ topi,
    float* __restrict__ mulv, int* __restrict__ muli) {
  extern __shared__ char smem[];
  float* rowbuf = (float*)smem;  // NDIRS floats (64KB, dynamic)
  unsigned long long* wred = (unsigned long long*)(smem + (size_t)NDIRS * sizeof(float));

  const int row  = blockIdx.x;
  const int t    = threadIdx.x;
  const int lane = t & 31;
  const int wid  = t >> 5;
  const float* src = lat_pre + (size_t)row * NDIRS;

  for (int j = t; j < NDIRS; j += 256) rowbuf[j] = src[j];
  __syncthreads();

  for (int i = 0; i < MULTIK; ++i) {
    unsigned long long best = 0ull;
    for (int j = t; j < NDIRS; j += 256) {
      unsigned u = __float_as_uint(rowbuf[j]);
      u ^= (u >> 31) ? 0xFFFFFFFFu : 0x80000000u;  // monotonic float key
      unsigned long long key = ((unsigned long long)u << 32) | (unsigned)(~j);
      if (key > best) best = key;
    }
#pragma unroll
    for (int off = 16; off > 0; off >>= 1) {  // wave32 reduction
      unsigned long long o = __shfl_xor(best, off, 32);
      if (o > best) best = o;
    }
    if (lane == 0) wred[wid] = best;
    __syncthreads();
    if (t == 0) {
      unsigned long long b = wred[0];
#pragma unroll
      for (int w = 1; w < 8; ++w)
        if (wred[w] > b) b = wred[w];
      const int idx = (int)(~(unsigned)(b & 0xFFFFFFFFull));  // lower index wins ties
      const float v  = rowbuf[idx];
      const float rv = v > 0.f ? v : 0.f;
      mulv[(size_t)row * MULTIK + i] = rv;
      muli[(size_t)row * MULTIK + i] = idx;
      if (i < TOPK) {
        topv[(size_t)row * TOPK + i] = rv;
        topi[(size_t)row * TOPK + i] = idx;
        latents[(size_t)row * NDIRS + idx] = rv;  // scatter into zeroed latents
      }
      rowbuf[idx] = -__builtin_inff();
    }
    __syncthreads();
  }
}

// -------------------- sparse decode: gather W_decT rows (L2-resident) -----------------
__global__ void __launch_bounds__(256) decode_kernel(
    const float* __restrict__ WdT, const float* __restrict__ pre_bias,
    const float* __restrict__ mulv, const int* __restrict__ muli,
    float* __restrict__ recons, float* __restrict__ mrecons) {
  __shared__ float sv[MULTIK];
  __shared__ int   si[MULTIK];
  const int row = blockIdx.x;
  const int t   = threadIdx.x;
  if (t < MULTIK) {
    sv[t] = mulv[(size_t)row * MULTIK + t];
    si[t] = muli[(size_t)row * MULTIK + t];
  }
  __syncthreads();

  float4 acc   = ((const float4*)pre_bias)[t];
  float4 acc32 = acc;
  for (int k = 0; k < MULTIK; ++k) {
    const float  v = sv[k];
    const float4 w = ((const float4*)(WdT + (size_t)si[k] * D_MODEL))[t];
    if (k + 1 < MULTIK)
      __builtin_prefetch(WdT + (size_t)si[k + 1] * D_MODEL + 4 * t, 0, 1);
    acc.x += v * w.x; acc.y += v * w.y; acc.z += v * w.z; acc.w += v * w.w;
    if (k == TOPK - 1) acc32 = acc;  // top-32 is the sorted prefix of top-128
  }
  ((float4*)(recons  + (size_t)row * D_MODEL))[t] = acc32;
  ((float4*)(mrecons + (size_t)row * D_MODEL))[t] = acc;
}

// ======================================================================================
extern "C" void kernel_launch(void* const* d_in, const int* in_sizes, int n_in,
                              void* d_out, int out_size, void* d_ws, size_t ws_size,
                              hipStream_t stream) {
  const float* x           = (const float*)d_in[0];
  const float* W_enc       = (const float*)d_in[1];
  const float* W_dec       = (const float*)d_in[2];
  const float* pre_bias    = (const float*)d_in[3];
  const float* latent_bias = (const float*)d_in[4];

  // Output segments, concatenated flat in reference return order.
  float* fout = (float*)d_out;
  size_t o = 0;
  float* recons  = fout + o; o += (size_t)N_ROWS * D_MODEL;
  float* mrecons = fout + o; o += (size_t)N_ROWS * D_MODEL;
  float* topv    = fout + o; o += (size_t)N_ROWS * TOPK;
  int*   topi    = (int*)(fout + o); o += (size_t)N_ROWS * TOPK;
  float* mulv    = fout + o; o += (size_t)N_ROWS * MULTIK;
  int*   muli    = (int*)(fout + o); o += (size_t)N_ROWS * MULTIK;
  float* lat_pre = fout + o; o += (size_t)N_ROWS * NDIRS;
  float* latents = fout + o;

  // Workspace: xbf (16MB) | wbf (32MB) | W_decT (64MB)
  char* ws = (char*)d_ws;
  __bf16* xbf = (__bf16*)ws;
  __bf16* wbf = (__bf16*)(ws + (size_t)N_ROWS * D_MODEL * 2);
  float*  WdT = (float*)(ws + (size_t)N_ROWS * D_MODEL * 2 + (size_t)NDIRS * D_MODEL * 2);

  hipLaunchKernelGGL(convert_x_kernel, dim3((N_ROWS * D_MODEL) / 256), dim3(256), 0, stream,
                     x, pre_bias, xbf);
  hipLaunchKernelGGL(convert_w_kernel, dim3((NDIRS * D_MODEL) / 256), dim3(256), 0, stream,
                     W_enc, wbf);
  hipLaunchKernelGGL(transpose_wdec_kernel, dim3(NDIRS / 32, D_MODEL / 32), dim3(32, 8), 0,
                     stream, W_dec, WdT);
  hipLaunchKernelGGL(zero_latents_kernel, dim3(((size_t)N_ROWS * NDIRS / 4) / 256), dim3(256),
                     0, stream, (float4*)latents);
  hipLaunchKernelGGL(encoder_gemm_kernel, dim3(NDIRS / 128, N_ROWS / 128), dim3(256), 0,
                     stream, xbf, wbf, latent_bias, lat_pre);
  hipLaunchKernelGGL(topk_kernel, dim3(N_ROWS), dim3(256),
                     (size_t)NDIRS * sizeof(float) + 8 * sizeof(unsigned long long), stream,
                     lat_pre, latents, topv, topi, mulv, muli);
  hipLaunchKernelGGL(decode_kernel, dim3(N_ROWS), dim3(256), 0, stream,
                     WdT, pre_bias, mulv, muli, recons, mrecons);
}
